// Attention_59270548685139
// MI455X (gfx1250) — matrix operand order repaired and checked by reading, not compiled, not run on hardware
//
#include <hip/hip_runtime.h>
#include <hip/hip_bf16.h>

#define B_     8
#define N_     1024
#define D_     1024
#define H_     8
#define DH_    64
#define INNER_ 512
#define SCALE_ 0.125f
#define EPS_   1e-7f

typedef __attribute__((ext_vector_type(16))) __bf16 v16bf;
typedef __attribute__((ext_vector_type(8)))  float  v8f;

// ---- fragment loaders ------------------------------------------------------
// A-matrix 16x32 bf16: lane l -> row m=l&15, kh=l>>4; K = {8kh+0..7, 8kh+16..23}
__device__ __forceinline__ v16bf load_frag_a(const __bf16* p) {
    union { uint4 u[2]; v16bf v; } t;
    t.u[0] = *(const uint4*)(p);        // K = 8kh + 0..7
    t.u[1] = *(const uint4*)(p + 16);   // K = 8kh + 16..23
    return t.v;
}
// B-matrix 32x16 bf16: lane l -> col n=l&15, kh=l>>4; K = {8kh+0..7, 8kh+16..23}
__device__ __forceinline__ v16bf load_frag_b(const __bf16* p) {
    union { uint4 u[2]; v16bf v; } t;
    t.u[0] = *(const uint4*)(p);        // K = 8kh + 0..7
    t.u[1] = *(const uint4*)(p + 16);   // K = 8kh + 16..23
    return t.v;
}

__device__ __forceinline__ v8f wmma_bf16(v16bf a, v16bf b, v8f c) {
    return __builtin_amdgcn_wmma_f32_16x16x32_bf16(
        false, a, false, b, (short)0, c, false, false);
}

__device__ __forceinline__ v8f vzero8() {
    v8f z = {0.f,0.f,0.f,0.f,0.f,0.f,0.f,0.f};
    return z;
}

// ---- prep kernels ----------------------------------------------------------
__global__ void cvt_bf16_kernel(const float* __restrict__ src,
                                __bf16* __restrict__ dst, int n) {
    int i = blockIdx.x * blockDim.x + threadIdx.x;
    if (i < n) dst[i] = (__bf16)src[i];
}

// src: [rows][cols] f32  ->  dst: [cols][rows] bf16
__global__ void transpose_cvt_kernel(const float* __restrict__ src,
                                     __bf16* __restrict__ dst,
                                     int rows, int cols) {
    int i = blockIdx.x * blockDim.x + threadIdx.x;
    if (i < rows * cols) {
        int r = i / cols, c = i % cols;
        dst[(size_t)c * rows + r] = (__bf16)src[i];
    }
}

// ---- tile helpers ----------------------------------------------------------
__device__ __forceinline__ void load_a2(v16bf (&a)[2], const __bf16* x0,
                                        const __bf16* x1, int kk, int kh) {
    a[0] = load_frag_a(x0 + kk + kh * 8);
    a[1] = load_frag_a(x1 + kk + kh * 8);
}
__device__ __forceinline__ void load_b4(v16bf (&b)[4], const __bf16* w0,
                                        size_t stride, int kk, int kh) {
    #pragma unroll
    for (int t = 0; t < 4; t++)
        b[t] = load_frag_b(w0 + (size_t)t * 16 * stride + kk + kh * 8);
}
__device__ __forceinline__ void mm8(v8f (&acc)[2][4], v16bf (&a)[2], v16bf (&b)[4]) {
    #pragma unroll
    for (int mi = 0; mi < 2; mi++)
        #pragma unroll
        for (int t = 0; t < 4; t++)
            acc[mi][t] = wmma_bf16(a[mi], b[t], acc[mi][t]);
}

// ---- QKV projection: [8192x1024] @ [1024x512] x3 ---------------------------
// wave tile: M=32, N=64 (8 accumulators), double-buffered K pipeline.
// grid = (256 Mtiles, 3 matrices), 8 waves/block split the 512 N columns.
__global__ __launch_bounds__(256) void qkv_kernel(
    const __bf16* __restrict__ x,   // [B*N][D]
    const __bf16* __restrict__ wt,  // [3][INNER][D]  (transposed weights)
    __bf16* __restrict__ qb,        // [b][h][n][dh]
    __bf16* __restrict__ kb,        // [b][h][n][dh]
    __bf16* __restrict__ vtb)       // [b][h][dh][n]  (pre-transposed V)
{
    const int lane = threadIdx.x & 31;
    const int wave = threadIdx.x >> 5;
    const int row  = lane & 15;
    const int kh   = lane >> 4;
    const int m0   = blockIdx.x * 32;
    const int which = blockIdx.y;
    const int nbase = wave * 64;
    const __bf16* w  = wt + (size_t)which * INNER_ * D_;
    const __bf16* x0 = x + (size_t)(m0 + row) * D_;
    const __bf16* x1 = x + (size_t)(m0 + 16 + row) * D_;
    const __bf16* w0 = w + (size_t)(nbase + row) * D_;

    v8f acc[2][4];
    #pragma unroll
    for (int mi = 0; mi < 2; mi++)
        #pragma unroll
        for (int t = 0; t < 4; t++) acc[mi][t] = vzero8();

    v16bf aP[2], bP[4], aQ[2], bQ[4];
    load_a2(aP, x0, x1, 0, kh);
    load_b4(bP, w0, D_, 0, kh);
    for (int kk = 0; kk < D_; kk += 64) {
        load_a2(aQ, x0, x1, kk + 32, kh);
        load_b4(bQ, w0, D_, kk + 32, kh);
        mm8(acc, aP, bP);
        if (kk + 64 < D_) {
            load_a2(aP, x0, x1, kk + 64, kh);
            load_b4(bP, w0, D_, kk + 64, kh);
        }
        mm8(acc, aQ, bQ);
    }

    #pragma unroll
    for (int mi = 0; mi < 2; mi++) {
        #pragma unroll
        for (int t = 0; t < 4; t++) {
            int col = nbase + t * 16 + row;   // 0..511
            int h = col >> 6, dh = col & 63;
            #pragma unroll
            for (int r = 0; r < 8; r++) {
                int m = m0 + mi * 16 + r + 8 * kh;
                int bidx = m >> 10, nidx = m & 1023;
                float v = acc[mi][t][r];
                if (which == 0)
                    qb[(((size_t)bidx * H_ + h) * N_ + nidx) * DH_ + dh] = (__bf16)v;
                else if (which == 1)
                    kb[(((size_t)bidx * H_ + h) * N_ + nidx) * DH_ + dh] = (__bf16)v;
                else
                    vtb[(((size_t)bidx * H_ + h) * DH_ + dh) * N_ + nidx] = (__bf16)v;
            }
        }
    }
}

// ---- pass 1: column (query-dim) softmax stats ------------------------------
// sim = q @ k^T * scale; C-frag: lane = column j, rows i = r + 8*kh.
// Online max/sum-exp over i per column j; stores max and RECIPROCAL sum.
__global__ __launch_bounds__(256) void colstats_kernel(
    const __bf16* __restrict__ qb, const __bf16* __restrict__ kb,
    float* __restrict__ cmax, float* __restrict__ crcp)
{
    const int lane = threadIdx.x & 31;
    const int wave = threadIdx.x >> 5;
    const int row  = lane & 15;
    const int kh   = lane >> 4;
    const int bh   = blockIdx.x;
    const int jb   = (blockIdx.y * 8 + wave) * 16;
    const __bf16* qh = qb + (size_t)bh * N_ * DH_;
    const __bf16* kp = kb + (size_t)bh * N_ * DH_;

    const v16bf kf0 = load_frag_b(kp + (size_t)(jb + row) * DH_ + 0 + kh * 8);
    const v16bf kf1 = load_frag_b(kp + (size_t)(jb + row) * DH_ + 32 + kh * 8);

    float rmax = -__builtin_inff();
    float rsum = 0.f;

    auto tile = [&](v16bf a0, v16bf a1) {
        v8f s = vzero8();
        s = wmma_bf16(a0, kf0, s);
        s = wmma_bf16(a1, kf1, s);
        float e[8];
        float tmax = -__builtin_inff();
        #pragma unroll
        for (int r = 0; r < 8; r++) { e[r] = s[r] * SCALE_; tmax = fmaxf(tmax, e[r]); }
        tmax = fmaxf(tmax, __shfl_xor(tmax, 16, 32));
        float nm = fmaxf(rmax, tmax);
        float ls = 0.f;
        #pragma unroll
        for (int r = 0; r < 8; r++) ls += __expf(e[r] - nm);
        ls += __shfl_xor(ls, 16, 32);
        rsum = rsum * __expf(rmax - nm) + ls;
        rmax = nm;
    };

    v16bf aP0, aP1, aQ0, aQ1;
    aP0 = load_frag_a(qh + (size_t)(0 + row) * DH_ + 0 + kh * 8);
    aP1 = load_frag_a(qh + (size_t)(0 + row) * DH_ + 32 + kh * 8);
    for (int it = 0; it < N_; it += 32) {
        aQ0 = load_frag_a(qh + (size_t)(it + 16 + row) * DH_ + 0 + kh * 8);
        aQ1 = load_frag_a(qh + (size_t)(it + 16 + row) * DH_ + 32 + kh * 8);
        tile(aP0, aP1);
        if (it + 32 < N_) {
            aP0 = load_frag_a(qh + (size_t)(it + 32 + row) * DH_ + 0 + kh * 8);
            aP1 = load_frag_a(qh + (size_t)(it + 32 + row) * DH_ + 32 + kh * 8);
        }
        tile(aQ0, aQ1);
    }
    if (lane < 16) {
        cmax[(size_t)bh * N_ + jb + lane] = rmax;
        crcp[(size_t)bh * N_ + jb + lane] = 1.0f / rsum;   // reciprocal for pass 2
    }
}

// ---- pass 2: P = exp(sim - cmax)*crcp; o = (P @ V) / (rowsum + eps) --------
// sim^T = K_rows @ Q^T so the C-frag (lane = query i, rows = key j) packs
// register-locally into the A-fragment of the P @ V WMMA. k-frags ping-pong.
__device__ __forceinline__ void load_ka4(v16bf (&d)[4], const __bf16* kp,
                                         int jt, int row, int kh) {
    d[0] = load_frag_a(kp + (size_t)(jt + row) * DH_ + 0 + kh * 8);
    d[1] = load_frag_a(kp + (size_t)(jt + row) * DH_ + 32 + kh * 8);
    d[2] = load_frag_a(kp + (size_t)(jt + 16 + row) * DH_ + 0 + kh * 8);
    d[3] = load_frag_a(kp + (size_t)(jt + 16 + row) * DH_ + 32 + kh * 8);
}

__global__ __launch_bounds__(256) void attnv_kernel(
    const __bf16* __restrict__ qb, const __bf16* __restrict__ kb,
    const __bf16* __restrict__ vtb, const float* __restrict__ cmax,
    const float* __restrict__ crcp, __bf16* __restrict__ ao)
{
    const int lane = threadIdx.x & 31;
    const int wave = threadIdx.x >> 5;
    const int row  = lane & 15;
    const int kh   = lane >> 4;
    const int bh   = blockIdx.x;
    const int bidx = bh >> 3, h = bh & 7;
    const int ib   = (blockIdx.y * 8 + wave) * 16;
    const __bf16* qh = qb + (size_t)bh * N_ * DH_;
    const __bf16* kp = kb + (size_t)bh * N_ * DH_;
    const __bf16* vp = vtb + (size_t)bh * DH_ * N_;
    const float* cmp = cmax + (size_t)bh * N_;
    const float* csp = crcp + (size_t)bh * N_;

    // Q^T as B-operand of sim^T (lane = query column i)
    const v16bf qf0 = load_frag_b(qh + (size_t)(ib + row) * DH_ + 0 + kh * 8);
    const v16bf qf1 = load_frag_b(qh + (size_t)(ib + row) * DH_ + 32 + kh * 8);

    v8f oacc[4];
    #pragma unroll
    for (int t = 0; t < 4; t++) oacc[t] = vzero8();
    float rs = 0.f;

    auto process = [&](int jc, int jn, v16bf (&kc)[4], v16bf (&kn)[4], bool pre) {
        // issue V fragments early: their latency hides behind sim WMMAs + exp
        v16bf vb0 = load_frag_b(vp + (size_t)(0 * 16 + row) * N_ + jc + kh * 8);
        v16bf vb1 = load_frag_b(vp + (size_t)(1 * 16 + row) * N_ + jc + kh * 8);
        v16bf vb2 = load_frag_b(vp + (size_t)(2 * 16 + row) * N_ + jc + kh * 8);
        v16bf vb3 = load_frag_b(vp + (size_t)(3 * 16 + row) * N_ + jc + kh * 8);
        v8f s0 = vzero8(), s1 = vzero8();
        s0 = wmma_bf16(kc[0], qf0, s0);
        s0 = wmma_bf16(kc[1], qf1, s0);
        s1 = wmma_bf16(kc[2], qf0, s1);
        s1 = wmma_bf16(kc[3], qf1, s1);
        if (pre) load_ka4(kn, kp, jn, row, kh);   // prefetch next j-block
        // pack probabilities: A-frag elem e<8 -> K=8kh+e ; e>=8 -> K=8kh+16+(e-8)
        v16bf pa;
        #pragma unroll
        for (int r = 0; r < 8; r++) {
            int j0 = jc + r + 8 * kh;
            float p0 = __expf(s0[r] * SCALE_ - cmp[j0]) * csp[j0];
            rs += p0;
            pa[r] = (__bf16)p0;
            int j1 = jc + 16 + r + 8 * kh;
            float p1 = __expf(s1[r] * SCALE_ - cmp[j1]) * csp[j1];
            rs += p1;
            pa[8 + r] = (__bf16)p1;
        }
        oacc[0] = wmma_bf16(pa, vb0, oacc[0]);
        oacc[1] = wmma_bf16(pa, vb1, oacc[1]);
        oacc[2] = wmma_bf16(pa, vb2, oacc[2]);
        oacc[3] = wmma_bf16(pa, vb3, oacc[3]);
    };

    v16bf kaP[4], kaQ[4];
    load_ka4(kaP, kp, 0, row, kh);
    for (int jb = 0; jb < N_; jb += 64) {
        process(jb,      jb + 32, kaP, kaQ, true);
        process(jb + 32, jb + 64, kaQ, kaP, jb + 64 < N_);
    }

    rs += __shfl_xor(rs, 16, 32);
    float inv = 1.f / (rs + EPS_);
    #pragma unroll
    for (int nt = 0; nt < 4; nt++) {
        #pragma unroll
        for (int r = 0; r < 8; r++) {
            float invr = __shfl(inv, r + 8 * kh, 32);   // lane (l&15)==i holds rowsum
            int i_g = ib + r + 8 * kh;
            int dh  = nt * 16 + row;
            ao[((size_t)bidx * N_ + i_g) * INNER_ + h * DH_ + dh] =
                (__bf16)(oacc[nt][r] * invr);
        }
    }
}

// ---- output projection: [8192x512] @ [512x1024] + bo -> f32 ----------------
// wave tile: M=32, N=64, double-buffered K pipeline. grid = (256, 2).
__global__ __launch_bounds__(256) void outproj_kernel(
    const __bf16* __restrict__ ao, const __bf16* __restrict__ wot,
    const float* __restrict__ bo, float* __restrict__ out)
{
    const int lane = threadIdx.x & 31;
    const int wave = threadIdx.x >> 5;
    const int row  = lane & 15;
    const int kh   = lane >> 4;
    const int m0   = blockIdx.x * 32;
    const int nbase = (blockIdx.y * 8 + wave) * 64;
    const __bf16* a0p = ao + (size_t)(m0 + row) * INNER_;
    const __bf16* a1p = ao + (size_t)(m0 + 16 + row) * INNER_;
    const __bf16* w0  = wot + (size_t)(nbase + row) * INNER_;

    v8f acc[2][4];
    #pragma unroll
    for (int mi = 0; mi < 2; mi++)
        #pragma unroll
        for (int t = 0; t < 4; t++) acc[mi][t] = vzero8();

    v16bf aP[2], bP[4], aQ[2], bQ[4];
    load_a2(aP, a0p, a1p, 0, kh);
    load_b4(bP, w0, INNER_, 0, kh);
    for (int kk = 0; kk < INNER_; kk += 64) {
        load_a2(aQ, a0p, a1p, kk + 32, kh);
        load_b4(bQ, w0, INNER_, kk + 32, kh);
        mm8(acc, aP, bP);
        if (kk + 64 < INNER_) {
            load_a2(aP, a0p, a1p, kk + 64, kh);
            load_b4(bP, w0, INNER_, kk + 64, kh);
        }
        mm8(acc, aQ, bQ);
    }

    #pragma unroll
    for (int mi = 0; mi < 2; mi++) {
        #pragma unroll
        for (int t = 0; t < 4; t++) {
            int col = nbase + t * 16 + row;
            float bias = bo[col];
            #pragma unroll
            for (int r = 0; r < 8; r++) {
                int m = m0 + mi * 16 + r + 8 * kh;
                out[(size_t)m * D_ + col] = acc[mi][t][r] + bias;
            }
        }
    }
}

// ---- launch ----------------------------------------------------------------
extern "C" void kernel_launch(void* const* d_in, const int* in_sizes, int n_in,
                              void* d_out, int out_size, void* d_ws, size_t ws_size,
                              hipStream_t stream) {
    const float* x  = (const float*)d_in[0];
    const float* Wq = (const float*)d_in[1];
    const float* Wk = (const float*)d_in[2];
    const float* Wv = (const float*)d_in[3];
    const float* Wo = (const float*)d_in[4];
    const float* bo = (const float*)d_in[5];
    float* out = (float*)d_out;

    char* ws = (char*)d_ws;
    size_t off = 0;
    auto wsAlloc = [&](size_t bytes) -> void* {
        void* p = ws + off;
        off += (bytes + 255) & ~(size_t)255;
        return p;
    };
    __bf16* xb  = (__bf16*)wsAlloc((size_t)B_ * N_ * D_ * 2);        // 16 MB
    __bf16* wt  = (__bf16*)wsAlloc((size_t)3 * INNER_ * D_ * 2);     //  3 MB
    __bf16* wot = (__bf16*)wsAlloc((size_t)D_ * INNER_ * 2);         //  1 MB
    __bf16* qb  = (__bf16*)wsAlloc((size_t)B_ * H_ * N_ * DH_ * 2);  //  8 MB
    __bf16* kb  = (__bf16*)wsAlloc((size_t)B_ * H_ * N_ * DH_ * 2);  //  8 MB
    __bf16* vtb = (__bf16*)wsAlloc((size_t)B_ * H_ * N_ * DH_ * 2);  //  8 MB
    float*  cm  = (float*)wsAlloc((size_t)B_ * H_ * N_ * 4);
    float*  cs  = (float*)wsAlloc((size_t)B_ * H_ * N_ * 4);
    __bf16* ao  = (__bf16*)wsAlloc((size_t)B_ * N_ * INNER_ * 2);    //  8 MB

    const int nx = B_ * N_ * D_;
    cvt_bf16_kernel<<<(nx + 255) / 256, 256, 0, stream>>>(x, xb, nx);
    const int nw = D_ * INNER_;
    transpose_cvt_kernel<<<(nw + 255) / 256, 256, 0, stream>>>(Wq, wt + 0 * (size_t)INNER_ * D_, D_, INNER_);
    transpose_cvt_kernel<<<(nw + 255) / 256, 256, 0, stream>>>(Wk, wt + 1 * (size_t)INNER_ * D_, D_, INNER_);
    transpose_cvt_kernel<<<(nw + 255) / 256, 256, 0, stream>>>(Wv, wt + 2 * (size_t)INNER_ * D_, D_, INNER_);
    transpose_cvt_kernel<<<(nw + 255) / 256, 256, 0, stream>>>(Wo, wot, INNER_, D_);

    qkv_kernel<<<dim3(256, 3), 256, 0, stream>>>(xb, wt, qb, kb, vtb);
    colstats_kernel<<<dim3(64, 8), 256, 0, stream>>>(qb, kb, cm, cs);
    attnv_kernel<<<dim3(64, 8), 256, 0, stream>>>(qb, kb, vtb, cm, cs, ao);
    outproj_kernel<<<dim3(256, 2), 256, 0, stream>>>(ao, wot, bo, out);
}
